// CarbonGNN_36447092474505
// MI455X (gfx1250) — matrix operand order repaired and checked by reading, not compiled, not run on hardware
//
#include <hip/hip_runtime.h>
#include <hip/hip_bf16.h>
#include <math.h>

typedef __attribute__((ext_vector_type(16))) __bf16 v16bf;
typedef __attribute__((ext_vector_type(8)))  __bf16 v8bf;
typedef __attribute__((ext_vector_type(8)))  float  v8f;

#define HID   128
#define HEADS 8
#define CPH   16

// ---------- monotonic float<->uint encode for atomicMax-based segment max ----------
__device__ __forceinline__ unsigned enc_f32(float f) {
  unsigned u = __float_as_uint(f);
  return (u & 0x80000000u) ? ~u : (u | 0x80000000u);
}
__device__ __forceinline__ float dec_f32(unsigned u) {
  unsigned b = (u & 0x80000000u) ? (u & 0x7FFFFFFFu) : ~u;
  return __uint_as_float(b);
}
#define ENC_NEG_INF 0x007FFFFFu   // enc(-inf)

// ---------- WMMA A fragment: row-major LDS tile, two contiguous 16B runs per lane ----------
// A (16x32, MxK): lanes 0-15 row M=lane hold K {0..7,16..23}; lanes 16-31 hold K {8..15,24..31}
__device__ __forceinline__ v16bf frag_A(const __bf16* As, int row, int lane) {
  int kb = (lane < 16) ? 0 : 8;
  const __bf16* p = As + row * 32 + kb;          // 16B aligned
  v8bf lo = *(const v8bf*)(p);
  v8bf hi = *(const v8bf*)(p + 16);
  return __builtin_shufflevector(lo, hi, 0,1,2,3,4,5,6,7,8,9,10,11,12,13,14,15);
}

// ---------- generic bf16 GEMM: out[M,128] = A[M,K] @ W[K,128] + bias, opt ReLU ----------
// B operand staged in fragment-ready swizzled LDS: each lane reads 32 contiguous bytes.
template <int K>
__global__ __launch_bounds__(256)
void k_gemm_bf16(const __bf16* __restrict__ A,
                 const float* __restrict__ W, const float* __restrict__ bias,
                 float* __restrict__ outf, __bf16* __restrict__ outb,
                 int M, int act) {
  constexpr int KT = K / 32;
  __shared__ __bf16 WsF[KT * 8 * 32 * 16];  // [ktile][coltile][lane][16] fragment-ready
  __shared__ __bf16 As[HID * 32];           // 128-row x 32-K A tile, row-major
  __shared__ float  bs[HID];

  const int tid  = threadIdx.x;
  const int wave = tid >> 5;
  const int lane = tid & 31;
  const int rowBase = blockIdx.x * 128;

  // Swizzle weights into fragment order (one-time cost per block).
  for (int idx = tid; idx < K * HID; idx += 256) {
    int k = idx >> 7, col = idx & 127;
    int kt = k >> 5, kk = k & 31;
    int ct = col >> 4, cc = col & 15;
    int ln = cc + ((kk & 8) << 1);      // lanes 0-15: K {0..7,16..23}, lanes 16-31: K {8..15,24..31}
    int i  = (kk & 7) | ((kk & 16) >> 1);
    WsF[(((kt * 8 + ct) * 32) + ln) * 16 + i] = (__bf16)W[idx];
  }
  if (tid < HID) bs[tid] = bias[tid];
  __syncthreads();

  v8f acc[8];
#pragma unroll
  for (int c = 0; c < 8; ++c) acc[c] = (v8f){0.f,0.f,0.f,0.f,0.f,0.f,0.f,0.f};

  const int rowLocal = wave * 16 + (lane & 15);

#pragma unroll
  for (int kt = 0; kt < KT; ++kt) {
    // cooperative 128x32 A-tile load: each thread moves 32 contiguous bytes
    {
      int r  = tid >> 1;                // 0..127
      int kc = (tid & 1) * 16;          // 0 or 16
      int gr = rowBase + r;
      __bf16* dst = As + r * 32 + kc;
      if (gr < M) {
        const __bf16* src = A + (size_t)gr * K + kt * 32 + kc;
        *(v8bf*)(dst)     = *(const v8bf*)(src);
        *(v8bf*)(dst + 8) = *(const v8bf*)(src + 8);
      } else {
#pragma unroll
        for (int i = 0; i < 16; ++i) dst[i] = (__bf16)0.0f;
      }
    }
    __syncthreads();
    v16bf af = frag_A(As, rowLocal, lane);
#pragma unroll
    for (int ct = 0; ct < 8; ++ct) {
      v16bf bf = *(const v16bf*)&WsF[(((kt * 8 + ct) * 32) + lane) * 16];
      acc[ct] = __builtin_amdgcn_wmma_f32_16x16x32_bf16(
          false, af, false, bf, (short)0, acc[ct], false, false);
    }
    __syncthreads();
  }

  const int rowHalf = (lane < 16) ? 0 : 8;
#pragma unroll
  for (int ct = 0; ct < 8; ++ct) {
    int col = ct * 16 + (lane & 15);
#pragma unroll
    for (int r = 0; r < 8; ++r) {
      int row = rowBase + wave * 16 + rowHalf + r;
      if (row < M) {
        float v = acc[ct][r] + bs[col];
        if (act == 1) v = (v > 0.f) ? v : 0.f;
        size_t o = (size_t)row * HID + col;
        if (outf) outf[o] = v;
        if (outb) outb[o] = (__bf16)v;
      }
    }
  }
}

// ---------- elementwise f32 -> bf16 cast ----------
__global__ void k_cast_bf16(const float* __restrict__ in, __bf16* __restrict__ out, int n) {
  int i = blockIdx.x * blockDim.x + threadIdx.x;
  int st = gridDim.x * blockDim.x;
  for (; i < n; i += st) out[i] = (__bf16)in[i];
}

// ---------- zero/init scratch ----------
__global__ void k_init(float* agg, float* ssum, unsigned* smax, float* easum, int N) {
  int i = blockIdx.x * blockDim.x + threadIdx.x;
  int st = gridDim.x * blockDim.x;
  int tot = N * HID;
  for (int j = i; j < tot; j += st) {
    agg[j] = 0.f;
    if (j < N * HEADS) { ssum[j] = 0.f; smax[j] = ENC_NEG_INF; }
    if (j < 16) easum[j] = 0.f;
  }
}

// ---------- mean of edge_attr over real edges (block-reduced) ----------
__global__ __launch_bounds__(256)
void k_ea_mean(const float* __restrict__ ea, float* __restrict__ easum, int E) {
  __shared__ float sm[16];
  if (threadIdx.x < 16) sm[threadIdx.x] = 0.f;
  __syncthreads();
  float loc[16];
#pragma unroll
  for (int j = 0; j < 16; ++j) loc[j] = 0.f;
  int i = blockIdx.x * blockDim.x + threadIdx.x;
  int st = gridDim.x * blockDim.x;
  for (int e = i; e < E; e += st) {
    const float* p = ea + (size_t)e * 16;
#pragma unroll
    for (int j = 0; j < 16; ++j) loc[j] += p[j];
  }
#pragma unroll
  for (int j = 0; j < 16; ++j) atomicAdd(&sm[j], loc[j]);
  __syncthreads();
  if (threadIdx.x < 16) atomicAdd(&easum[threadIdx.x], sm[threadIdx.x]);
}

// ---------- edge pass 1: GATv2 score + segment max (W_e projection fused, never materialized) ----------
__global__ __launch_bounds__(256)
void k_edge_score(const float* __restrict__ ea, const int* __restrict__ ei,
                  const float* __restrict__ We, const float* __restrict__ att,
                  const float* __restrict__ easum,
                  const float* __restrict__ xl, const float* __restrict__ xr,
                  float* __restrict__ scores, unsigned* __restrict__ smax,
                  int E, int N) {
  __shared__ float We_s[16 * HID];
  __shared__ float att_s[HID];
  for (int i = threadIdx.x; i < 16 * HID; i += 256) We_s[i] = We[i];
  if (threadIdx.x < HID) att_s[threadIdx.x] = att[threadIdx.x];
  __syncthreads();

  int tid = blockIdx.x * 256 + threadIdx.x;
  int Et = E + N;
  if (tid >= Et * HEADS) return;
  int e = tid >> 3, h = tid & 7;
  int s, d;
  float eav[16];
  if (e < E) {
    s = ei[e]; d = ei[E + e];
    const float* p = ea + (size_t)e * 16;
#pragma unroll
    for (int j = 0; j < 16; ++j) eav[j] = p[j];
  } else {
    int i = e - E; s = i; d = i;
    float invE = 1.0f / (float)E;
#pragma unroll
    for (int j = 0; j < 16; ++j) eav[j] = easum[j] * invE;   // fill_value='mean'
  }
  const float* xlp = xl + (size_t)s * HID + h * CPH;
  const float* xrp = xr + (size_t)d * HID + h * CPH;
  const float* ap  = att_s + h * CPH;
  float sc = 0.f;
#pragma unroll
  for (int c = 0; c < CPH; ++c) {
    float w = 0.f;
#pragma unroll
    for (int j = 0; j < 16; ++j) w += eav[j] * We_s[j * HID + h * CPH + c];
    float m = xlp[c] + xrp[c] + w;
    m = (m > 0.f) ? m : 0.2f * m;    // leaky_relu(0.2)
    sc += ap[c] * m;
  }
  scores[tid] = sc;
  atomicMax(&smax[d * HEADS + h], enc_f32(sc));
}

// ---------- edge pass 2: exp(score - max) + segment sum ----------
__global__ __launch_bounds__(256)
void k_edge_exp(const int* __restrict__ ei, const unsigned* __restrict__ smax,
                float* __restrict__ scores, float* __restrict__ ssum, int E, int N) {
  int tid = blockIdx.x * 256 + threadIdx.x;
  int Et = E + N;
  if (tid >= Et * HEADS) return;
  int e = tid >> 3, h = tid & 7;
  int d = (e < E) ? ei[E + e] : (e - E);
  float mx = dec_f32(smax[d * HEADS + h]);
  float ex = __expf(scores[tid] - mx);
  scores[tid] = ex;
  atomicAdd(&ssum[d * HEADS + h], ex);
}

// ---------- edge pass 3: alpha-weighted scatter-add of xl[src] ----------
__global__ __launch_bounds__(256)
void k_edge_agg(const int* __restrict__ ei, const float* __restrict__ scores,
                const float* __restrict__ ssum, const float* __restrict__ xl,
                float* __restrict__ agg, int E, int N) {
  int tid = blockIdx.x * 256 + threadIdx.x;
  int Et = E + N;
  if (tid >= Et * HEADS) return;
  int e = tid >> 3, h = tid & 7;
  int s = (e < E) ? ei[e]     : (e - E);
  int d = (e < E) ? ei[E + e] : (e - E);
  float alpha = scores[tid] / (ssum[d * HEADS + h] + 1e-16f);
  const float* xlp = xl + (size_t)s * HID + h * CPH;
  float* ag = agg + (size_t)d * HID + h * CPH;
#pragma unroll
  for (int c = 0; c < CPH; ++c) atomicAdd(&ag[c], alpha * xlp[c]);
}

// ---------- epilogue: bias + leaky(0.01) + residual + LayerNorm (one wave32 per node) ----------
__global__ __launch_bounds__(256)
void k_node_ln(const float* __restrict__ agg, const float* __restrict__ b_conv,
               const float* __restrict__ x0, const float* __restrict__ ln_w,
               const float* __restrict__ ln_b, __bf16* __restrict__ zb, int N) {
  int node = blockIdx.x * 8 + (threadIdx.x >> 5);
  int lane = threadIdx.x & 31;
  if (node >= N) return;
  const float* ag  = agg + (size_t)node * HID;
  const float* x0p = x0  + (size_t)node * HID;
  float vals[4], s = 0.f, s2 = 0.f;
#pragma unroll
  for (int i = 0; i < 4; ++i) {
    int c = lane * 4 + i;
    float v = ag[c] + b_conv[c];
    v = (v > 0.f) ? v : 0.01f * v;
    v += x0p[c];
    vals[i] = v; s += v; s2 += v * v;
  }
#pragma unroll
  for (int off = 16; off >= 1; off >>= 1) {   // wave32 reduction
    s  += __shfl_xor(s, off, 32);
    s2 += __shfl_xor(s2, off, 32);
  }
  float mu  = s * (1.f / HID);
  float var = s2 * (1.f / HID) - mu * mu;
  float rs  = rsqrtf(var + 1e-5f);
  __bf16* zp = zb + (size_t)node * HID;
#pragma unroll
  for (int i = 0; i < 4; ++i) {
    int c = lane * 4 + i;
    zp[c] = (__bf16)((vals[i] - mu) * rs * ln_w[c] + ln_b[c]);
  }
}

// ---------- final dot with W2 + softplus ----------
__global__ void k_out(const float* __restrict__ t, const float* __restrict__ W2,
                      const float* __restrict__ b2, float* __restrict__ out, int N) {
  int i = blockIdx.x * blockDim.x + threadIdx.x;
  if (i >= N) return;
  const float* tp = t + (size_t)i * HID;
  float acc = b2[0];
#pragma unroll 8
  for (int c = 0; c < HID; ++c) acc += tp[c] * W2[c];
  out[i] = (acc > 0.f) ? (acc + log1pf(__expf(-acc))) : log1pf(__expf(acc));
}

extern "C" void kernel_launch(void* const* d_in, const int* in_sizes, int n_in,
                              void* d_out, int out_size, void* d_ws, size_t ws_size,
                              hipStream_t stream) {
  const float* x    = (const float*)d_in[0];
  const int*   ei   = (const int*)d_in[1];
  const float* ea   = (const float*)d_in[2];
  const float* W_fc = (const float*)d_in[3];
  const float* b_fc = (const float*)d_in[4];
  const float* W_l  = (const float*)d_in[5];
  const float* b_l  = (const float*)d_in[6];
  const float* W_r  = (const float*)d_in[7];
  const float* b_r  = (const float*)d_in[8];
  const float* W_e  = (const float*)d_in[9];
  const float* att  = (const float*)d_in[10];
  const float* b_cv = (const float*)d_in[11];
  const float* ln_w = (const float*)d_in[12];
  const float* ln_b = (const float*)d_in[13];
  const float* W1   = (const float*)d_in[14];
  const float* b1   = (const float*)d_in[15];
  const float* W2   = (const float*)d_in[16];
  const float* b2   = (const float*)d_in[17];
  float* out = (float*)d_out;

  const int N  = in_sizes[0] / 64;
  const int E  = in_sizes[1] / 2;
  const int Et = E + N;

  char* base = (char*)d_ws;
  size_t off = 0;
  auto carve = [&](size_t bytes) -> char* {
    off = (off + 255) & ~(size_t)255;
    char* p = base + off;
    off += bytes;
    return p;
  };
  __bf16*   xb     = (__bf16*)  carve((size_t)N * 64 * 2);
  float*    x0     = (float*)   carve((size_t)N * HID * 4);
  __bf16*   x0b    = (__bf16*)  carve((size_t)N * HID * 2);
  float*    xl     = (float*)   carve((size_t)N * HID * 4);
  float*    xr     = (float*)   carve((size_t)N * HID * 4);
  float*    scores = (float*)   carve((size_t)Et * HEADS * 4);
  unsigned* smax   = (unsigned*)carve((size_t)N * HEADS * 4);
  float*    ssum   = (float*)   carve((size_t)N * HEADS * 4);
  float*    agg    = (float*)   carve((size_t)N * HID * 4);
  __bf16*   zb     = (__bf16*)  carve((size_t)N * HID * 2);
  float*    t      = (float*)   carve((size_t)N * HID * 4);
  float*    easum  = (float*)   carve(16 * 4);
  (void)ws_size; (void)n_in; (void)out_size;

  const int gemmGrid = (N + 127) / 128;
  const int edgeGrid = (Et * HEADS + 255) / 256;
  const int gsGrid   = 2048;   // grid-stride kernels

  k_cast_bf16<<<gsGrid, 256, 0, stream>>>(x, xb, N * 64);
  k_gemm_bf16<64><<<gemmGrid, 256, 0, stream>>>(xb,  W_fc, b_fc, x0, x0b, N, 0);
  k_gemm_bf16<128><<<gemmGrid, 256, 0, stream>>>(x0b, W_l, b_l, xl, nullptr, N, 0);
  k_gemm_bf16<128><<<gemmGrid, 256, 0, stream>>>(x0b, W_r, b_r, xr, nullptr, N, 0);
  k_init<<<gsGrid, 256, 0, stream>>>(agg, ssum, smax, easum, N);
  k_ea_mean<<<gsGrid, 256, 0, stream>>>(ea, easum, E);
  k_edge_score<<<edgeGrid, 256, 0, stream>>>(ea, ei, W_e, att, easum, xl, xr, scores, smax, E, N);
  k_edge_exp<<<edgeGrid, 256, 0, stream>>>(ei, smax, scores, ssum, E, N);
  k_edge_agg<<<edgeGrid, 256, 0, stream>>>(ei, scores, ssum, xl, agg, E, N);
  k_node_ln<<<(N + 7) / 8, 256, 0, stream>>>(agg, b_cv, x0, ln_w, ln_b, zb, N);
  k_gemm_bf16<128><<<gemmGrid, 256, 0, stream>>>(zb, W1, b1, t, nullptr, N, 1);
  k_out<<<(N + 255) / 256, 256, 0, stream>>>(t, W2, b2, out, N);
}
